// BahdanauAttention_5695126634620
// MI455X (gfx1250) — compile-verified, matrix operand order checked by reading, not run on hardware
//
#include <hip/hip_runtime.h>

// ---------------------------------------------------------------------------
// Bahdanau attention on MI455X (gfx1250, wave32, WMMA).
// B=64, S=2048, D=E=1024.
//
// k_proj GEMM = 275 GFLOP over 512 MB of keys (1 HBM pass = 22us @ 23.3TB/s)
// -> compute bound -> bf16 WMMA (v_wmma_f32_16x16x32_bf16, f32 accumulate).
// Fuse +q_proj+W2_b, hardware v_tanh_f32, dot-with-V and the row reduction so
// the [B,S,D] intermediate never leaves registers.  S_TILE=64 keys tile held
// as pre-swizzled WMMA-A fragments in 128 KB LDS (CDNA5 allows 320 KB/WG),
// halving W2 fragment replay traffic from L2 vs a 32-row tile.
// V_b shifts all logits equally -> cancels in softmax -> dropped.
// ---------------------------------------------------------------------------

typedef __attribute__((ext_vector_type(16))) __bf16 v16bf;
typedef __attribute__((ext_vector_type(8)))  float  v8f;

#define Bsz 64
#define Ssz 2048
#define Dsz 1024
#define Esz 1024
#define S_TILE 64            // rows of keys per workgroup
#define NSB (S_TILE / 16)    // 4 row sub-blocks of 16
#define NKT (Esz / 32)       // 32 K-tiles of 32

__device__ __forceinline__ unsigned f2bf(float f) {
  unsigned u = __float_as_uint(f);
  u += 0x7FFFu + ((u >> 16) & 1u);          // round-to-nearest-even
  return u >> 16;
}

__device__ __forceinline__ float fast_tanh(float x) {
#if __has_builtin(__builtin_amdgcn_tanhf)
  return __builtin_amdgcn_tanhf(x);
#else
  float r;
  asm volatile("v_tanh_f32 %0, %1\n\tv_nop\n\tv_nop" : "=v"(r) : "v"(x));
  return r;
#endif
}

// --- pack W2 (f32 [D,E]) into bf16 wave32 WMMA-B (32x16) fragments ---------
// out[((dtile*NKT + kt)*32 + lane)*16 + e]
//   n  = lane & 15          (B-matrix column -> d = dtile*16 + n)
//   k_local = (e>>3)*16 + (lane>>4)*8 + (e&7)   (ISA 16-bit B layout)
__global__ void k_convert_w2(const float* __restrict__ W2,
                             unsigned short* __restrict__ out) {
  unsigned idx = blockIdx.x * 256u + threadIdx.x;   // 2^20 elements
  int e     = idx & 15;
  int lane  = (idx >> 4) & 31;
  int kt    = (idx >> 9) & 31;
  int dtile = idx >> 14;
  int n  = lane & 15;
  int kl = ((e >> 3) & 1) * 16 + (lane >> 4) * 8 + (e & 7);
  int d  = dtile * 16 + n;
  int k  = kt * 32 + kl;
  out[idx] = (unsigned short)f2bf(W2[(size_t)d * Esz + k]);
}

// --- q_proj[b,d] = query[b,:]·W1[d,:] + W1_b[d] + W2_b[d] ------------------
__global__ void k_qproj(const float* __restrict__ query,
                        const float* __restrict__ W1w,
                        const float* __restrict__ W1b,
                        const float* __restrict__ W2b,
                        float* __restrict__ qp) {
  int idx = blockIdx.x * 256 + threadIdx.x;
  int b = idx & (Bsz - 1);
  int d = idx >> 6;
  const float4* qr = (const float4*)(query + (size_t)b * Dsz);
  const float4* wr = (const float4*)(W1w + (size_t)d * Dsz);
  float acc = 0.f;
#pragma unroll 4
  for (int i = 0; i < Dsz / 4; ++i) {
    float4 q = qr[i], w = wr[i];
    acc += q.x * w.x + q.y * w.y + q.z * w.z + q.w * w.w;
  }
  qp[(size_t)b * Dsz + d] = acc + W1b[d] + W2b[d];
}

// --- fused scores kernel ---------------------------------------------------
// One WG (256 thr = 8 waves) per (b, 64-row s-tile).  Stage keys tile as
// bf16 WMMA-A fragments in LDS (128 KB); each wave owns 16 d-columns per
// iteration: kp(64x16) via 4x32 WMMA accumulations over K=1024, then
// score += V[d]*tanh(qp[b,d]+kp).  Cross-lane/-wave reduce, write scores.
__global__ void __launch_bounds__(256)
k_scores(const float* __restrict__ keys,
         const __bf16* __restrict__ w2b,
         const float* __restrict__ qp,
         const float* __restrict__ Vw,
         float* __restrict__ scores) {
  __shared__ __align__(32) __bf16 ldsK[NSB * NKT * 32 * 16];   // 128 KB

  int blk = blockIdx.x;
  int b   = blk >> 5;                 // 32 s-tiles per batch row
  int st  = blk & 31;
  int t   = threadIdx.x;
  int w   = t >> 5;                   // wave id 0..7
  int l   = t & 31;                   // lane id

  // ---- stage 1: keys[b, st*64 .. +63, :] -> bf16 A-fragments in LDS ------
  // Process octets of 8 consecutive e (e%8==0): same fragment lane, 8
  // consecutive fragment elements -> one 16B LDS store per 2 float4 loads.
  const float4* kb4 = (const float4*)(keys + ((size_t)b * Ssz + st * S_TILE) * Esz);
  for (int i = 0; i < (S_TILE * Esz / 8) / 256; ++i) {   // 32 octets / thread
    int p     = i * 256 + t;                 // octet index
    int s_row = p >> 7;                      // 128 octets per row
    int e0    = (p & 127) * 8;
    float4 f0 = kb4[(size_t)s_row * (Esz / 4) + (p & 127) * 2];
    float4 f1 = kb4[(size_t)s_row * (Esz / 4) + (p & 127) * 2 + 1];
    uint4 pk;
    pk.x = f2bf(f0.x) | (f2bf(f0.y) << 16);
    pk.y = f2bf(f0.z) | (f2bf(f0.w) << 16);
    pk.z = f2bf(f1.x) | (f2bf(f1.y) << 16);
    pk.w = f2bf(f1.z) | (f2bf(f1.w) << 16);
    int sb = s_row >> 4, m = s_row & 15;
    int kt = e0 >> 5;
    int lane  = (((e0 >> 3) & 1) << 4) + m;
    int eidx0 = ((e0 >> 4) & 1) << 3;        // 16B-aligned fragment offset
    *(uint4*)((char*)ldsK + ((((sb * NKT + kt) * 32 + lane) * 16 + eidx0) << 1)) = pk;
  }
  __syncthreads();

  // ---- stage 2: WMMA GEMM + fused tanh/V-dot -----------------------------
  const v16bf* afr = (const v16bf*)ldsK;
  const v16bf* bfr = (const v16bf*)w2b;

  float slot[NSB][8];
#pragma unroll
  for (int j = 0; j < NSB; ++j)
#pragma unroll
    for (int r = 0; r < 8; ++r) slot[j][r] = 0.f;

  for (int it = 0; it < 8; ++it) {
    int dtile = it * 8 + w;
    int d     = dtile * 16 + (l & 15);
    float qv  = qp[(size_t)b * Dsz + d];
    float vw  = Vw[d];

    v8f acc[NSB];
#pragma unroll
    for (int j = 0; j < NSB; ++j) acc[j] = (v8f){};

#pragma unroll 2
    for (int kt = 0; kt < NKT; ++kt) {
      v16bf bb = bfr[((size_t)dtile * NKT + kt) * 32 + l];
#pragma unroll
      for (int j = 0; j < NSB; ++j) {
        v16bf a = afr[(j * NKT + kt) * 32 + l];
        acc[j] = __builtin_amdgcn_wmma_f32_16x16x32_bf16(false, a, false, bb,
                                                         (short)0, acc[j],
                                                         false, false);
      }
    }
#pragma unroll
    for (int j = 0; j < NSB; ++j)
#pragma unroll
      for (int r = 0; r < 8; ++r)
        slot[j][r] += vw * fast_tanh(qv + acc[j][r]);
  }

  // ---- stage 3: reduce 16-lane groups, then waves via LDS (reuse ldsK) ---
  __syncthreads();                       // everyone done reading keys LDS
  float* sAcc = (float*)ldsK;
  if (t < S_TILE) sAcc[t] = 0.f;
  __syncthreads();

#pragma unroll
  for (int j = 0; j < NSB; ++j)
#pragma unroll
    for (int r = 0; r < 8; ++r) {
      float v = slot[j][r];
#pragma unroll
      for (int off = 8; off > 0; off >>= 1) v += __shfl_xor(v, off, 32);
      if ((l & 15) == 0)
        atomicAdd(&sAcc[j * 16 + r + 8 * (l >> 4)], v);
    }
  __syncthreads();
  if (t < S_TILE)
    scores[(size_t)b * Ssz + st * S_TILE + t] = sAcc[t];
}

// --- softmax over S per batch row ------------------------------------------
__global__ void k_softmax(const float* __restrict__ scores,
                          float* __restrict__ attn) {
  int b = blockIdx.x, t = threadIdx.x;
  __shared__ float red[256];
  const float* row = scores + (size_t)b * Ssz;
  float m = -3.4e38f;
  for (int s = t; s < Ssz; s += 256) m = fmaxf(m, row[s]);
  red[t] = m; __syncthreads();
  for (int o = 128; o > 0; o >>= 1) { if (t < o) red[t] = fmaxf(red[t], red[t + o]); __syncthreads(); }
  m = red[0]; __syncthreads();
  float sum = 0.f;
  for (int s = t; s < Ssz; s += 256) sum += __expf(row[s] - m);
  red[t] = sum; __syncthreads();
  for (int o = 128; o > 0; o >>= 1) { if (t < o) red[t] += red[t + o]; __syncthreads(); }
  float inv = 1.0f / red[0];
  for (int s = t; s < Ssz; s += 256)
    attn[(size_t)b * Ssz + s] = __expf(row[s] - m) * inv;
}

// --- context[b,e] = sum_s attn[b,s] * keys[b,s,e] --------------------------
__global__ void k_context(const float* __restrict__ attn,
                          const float* __restrict__ keys,
                          float* __restrict__ ctx) {
  int b  = blockIdx.x >> 3;
  int e  = (blockIdx.x & 7) * 128 + threadIdx.x;
  const float* kb = keys + (size_t)b * Ssz * Esz;
  const float* ab = attn + (size_t)b * Ssz;
  float acc = 0.f;
#pragma unroll 4
  for (int s = 0; s < Ssz; ++s) acc += ab[s] * kb[(size_t)s * Esz + e];
  ctx[(size_t)b * Esz + e] = acc;
}

extern "C" void kernel_launch(void* const* d_in, const int* in_sizes, int n_in,
                              void* d_out, int out_size, void* d_ws, size_t ws_size,
                              hipStream_t stream) {
  const float* query = (const float*)d_in[0];
  const float* keys  = (const float*)d_in[1];
  const float* W1w   = (const float*)d_in[2];
  const float* W1b   = (const float*)d_in[3];
  const float* W2w   = (const float*)d_in[4];
  const float* W2b   = (const float*)d_in[5];
  const float* Vw    = (const float*)d_in[6];
  // d_in[7] = V_b: constant shift of all logits -> cancels in softmax.

  // workspace layout
  unsigned short* w2frag = (unsigned short*)d_ws;                       // 2 MB
  float* qp     = (float*)((char*)d_ws + (size_t)2 * 1024 * 1024);      // 256 KB
  float* scores = (float*)((char*)d_ws + (size_t)2 * 1024 * 1024 + 256 * 1024); // 512 KB

  float* ctx  = (float*)d_out;                   // [64,1024]
  float* attn = (float*)d_out + Bsz * Esz;       // [64,2048]

  k_convert_w2<<<(Dsz * Esz) / 256, 256, 0, stream>>>(W2w, w2frag);
  k_qproj     <<<(Bsz * Dsz) / 256, 256, 0, stream>>>(query, W1w, W1b, W2b, qp);
  k_scores    <<<Bsz * (Ssz / S_TILE), 256, 0, stream>>>(keys, (const __bf16*)w2frag,
                                                         qp, Vw, scores);
  k_softmax   <<<Bsz, 256, 0, stream>>>(scores, attn);
  k_context   <<<Bsz * 8, 128, 0, stream>>>(attn, keys, ctx);
}
